// ScaledDotProductAttention_87153476370866
// MI455X (gfx1250) — matrix operand order repaired and checked by reading, not compiled, not run on hardware
//
#include <hip/hip_runtime.h>
#include <hip/hip_bf16.h>

// CDNA5 / gfx1250 attention producing (context, attn) as the reference does.
// bf16 WMMA (v_wmma_f32_16x16x32_bf16) with fp32 accumulate + fp32 softmax.
// attn (1.07GB) is the bandwidth floor (~46us @ 23.3 TB/s); bf16 WMMA keeps
// compute (~30us) under it. attn is written with non-temporal b128 stores so
// the 192MB L2 stays dedicated to the Q/K/V working set (48MB, L2-resident).

typedef __attribute__((ext_vector_type(16))) __bf16 v16bf;
typedef __attribute__((ext_vector_type(8)))  float  v8f;
typedef __attribute__((ext_vector_type(4)))  float  v4f;

#define BATCH 16
#define SEQ   4096
#define HDIM  64
#define QTILE 16
#define NWAVES 16          // 512 threads, wave32
#define BLOCK (NWAVES * 32)

// LDS: fp32 scores [QTILE][SEQ] (256KB) + P@V partials [16 waves][16][16] (16KB)
#define SC_BYTES   (QTILE * SEQ * sizeof(float))
#define PART_BYTES (NWAVES * 16 * 16 * sizeof(float))

__global__ __launch_bounds__(BLOCK) void
sdpa_wmma_kernel(const float* __restrict__ Q, const float* __restrict__ K,
                 const float* __restrict__ V, const unsigned char* __restrict__ mask,
                 float* __restrict__ ctx_out, float* __restrict__ attn_out)
{
    extern __shared__ char smem_raw[];
    float* sc   = reinterpret_cast<float*>(smem_raw);                 // [QTILE][SEQ]
    float* part = reinterpret_cast<float*>(smem_raw + SC_BYTES);      // [NWAVES][16][16]

    const int wg   = blockIdx.x;
    const int b    = wg / (SEQ / QTILE);
    const int qb   = (wg % (SEQ / QTILE)) * QTILE;
    const int tid  = threadIdx.x;
    const int wave = tid >> 5;
    const int lane = tid & 31;
    const int lhalf = lane >> 4;   // 0 | 1 : lane half selects K sub-range
    const int lmod  = lane & 15;

    const float* Qb = Q + (size_t)b * SEQ * HDIM;
    const float* Kb = K + (size_t)b * SEQ * HDIM;
    const float* Vb = V + (size_t)b * SEQ * HDIM;

    const float scale = 0.125f;    // 1/sqrt(64)

    // ---------------- Phase 1: scores = scale * Q K^T into LDS ----------------
    // A fragment (16x32 bf16, ISA layout): lane holds row m = lmod;
    // element j -> k = (j&7) + ((j>>3)<<4) + (lhalf<<3), per head-dim chunk.
    v16bf a0, a1;
    {
        const float* qrow = Qb + (size_t)(qb + lmod) * HDIM;
#pragma unroll
        for (int j = 0; j < 16; ++j) {
            const int koff = (j & 7) + ((j >> 3) << 4) + (lhalf << 3);
            a0[j] = (__bf16)(qrow[koff]      * scale);
            a1[j] = (__bf16)(qrow[32 + koff] * scale);
        }
    }
    // Each wave covers key columns [wave*256, wave*256+256): 16 tiles of 16.
    for (int t = 0; t < 16; ++t) {
        const int kb = wave * 256 + t * 16;
        // B fragment (32x16): B[k][n] = K[kb+n][k]; lane holds column n = lmod,
        // element j -> k = (j&7) + ((j>>3)<<4) + (lhalf<<3) within each 32-chunk.
        v16bf b0, b1;
        const float* krow = Kb + (size_t)(kb + lmod) * HDIM;
#pragma unroll
        for (int j = 0; j < 16; ++j) {
            const int kk = (j & 7) + ((j >> 3) << 4) + (lhalf << 3);
            b0[j] = (__bf16)krow[kk];
            b1[j] = (__bf16)krow[32 + kk];
        }
        v8f c = {};
        c = __builtin_amdgcn_wmma_f32_16x16x32_bf16(false, a0, false, b0,
                                                    (short)0, c, false, false);
        c = __builtin_amdgcn_wmma_f32_16x16x32_bf16(false, a1, false, b1,
                                                    (short)0, c, false, false);
        // D layout: VGPR r, lane -> M = r + 8*lhalf, N = lmod
#pragma unroll
        for (int r = 0; r < 8; ++r)
            sc[(r + 8 * lhalf) * SEQ + kb + lmod] = c[r];
    }
    __syncthreads();

    // ---------------- Phase 2: masked softmax, one wave per query row ----------
    // 4-wide vectorized row scan: ds_load_b128 + 4-byte mask words, branch-free
    // selects (no exec-masked LDS loads), non-temporal b128 attn stores.
    {
        const int row = wave;                       // 16 waves <-> 16 rows
        const unsigned int* mrow32 =
            (const unsigned int*)(mask + (size_t)b * SEQ);
        float* srow = sc + (size_t)row * SEQ;
        v4f* srow4 = (v4f*)srow;
        const int NC = SEQ / 4;                     // 1024 float4 chunks

        float mx = -INFINITY;
        for (int c = lane; c < NC; c += 32) {
            v4f s = srow4[c];                       // unconditional b128 load
            const unsigned int m = mrow32[c];
            s.x = (m & 0x000000ffu) ? -INFINITY : s.x;
            s.y = (m & 0x0000ff00u) ? -INFINITY : s.y;
            s.z = (m & 0x00ff0000u) ? -INFINITY : s.z;
            s.w = (m & 0xff000000u) ? -INFINITY : s.w;
            srow4[c] = s;
            mx = fmaxf(mx, fmaxf(fmaxf(s.x, s.y), fmaxf(s.z, s.w)));
        }
#pragma unroll
        for (int off = 16; off > 0; off >>= 1)
            mx = fmaxf(mx, __shfl_xor(mx, off, 32));

        float sum = 0.f;
        for (int c = lane; c < NC; c += 32) {
            v4f s = srow4[c];
            s.x = __expf(s.x - mx);
            s.y = __expf(s.y - mx);
            s.z = __expf(s.z - mx);
            s.w = __expf(s.w - mx);
            srow4[c] = s;
            sum += (s.x + s.y) + (s.z + s.w);
        }
#pragma unroll
        for (int off = 16; off > 0; off >>= 1)
            sum += __shfl_xor(sum, off, 32);
        const float inv = (sum > 0.f) ? (1.f / sum) : 0.f;

        v4f* arow4 = (v4f*)(attn_out + ((size_t)b * SEQ + (qb + row)) * SEQ);
        for (int c = lane; c < NC; c += 32) {
            v4f s = srow4[c];
            s.x *= inv; s.y *= inv; s.z *= inv; s.w *= inv;
            srow4[c] = s;                           // normalized P for P@V
            __builtin_nontemporal_store(s, arow4 + c);  // NT: don't pollute L2
        }
    }
    __syncthreads();

    // ---------------- Phase 3: context = P @ V (bf16 WMMA) --------------------
    // wave -> (n-tile = wave&3, key-quarter = wave>>2 covering 1024 keys)
    {
        const int nb = (wave & 3) * 16;
        const int k0 = (wave >> 2) * 1024;
        v8f c = {};
        for (int ch = 0; ch < 32; ++ch) {
            const int kb = k0 + ch * 32;
            // A from LDS P: row m = lmod; elem j -> k = kb + (j&7)+((j>>3)<<4)+(lhalf<<3)
            v16bf a;
            const float* prow = sc + (size_t)lmod * SEQ + kb;
#pragma unroll
            for (int j = 0; j < 16; ++j) {
                const int koff = (j & 7) + ((j >> 3) << 4) + (lhalf << 3);
                a[j] = (__bf16)prow[koff];
            }
            // B from V: B[k][n] = V[kb+k][nb+n]; lane n = lmod; elem j -> k = (j&7)+((j>>3)<<4)+(lhalf<<3)
            v16bf bb;
#pragma unroll
            for (int j = 0; j < 16; ++j) {
                const int kk = (j & 7) + ((j >> 3) << 4) + (lhalf << 3);
                bb[j] = (__bf16)Vb[(size_t)(kb + kk) * HDIM + nb + lmod];
            }
            c = __builtin_amdgcn_wmma_f32_16x16x32_bf16(false, a, false, bb,
                                                        (short)0, c, false, false);
        }
        // partial[wave][M][N], M = r + 8*lhalf, N = lmod
        float* pw = part + wave * 256;
#pragma unroll
        for (int r = 0; r < 8; ++r)
            pw[(r + 8 * lhalf) * 16 + lmod] = c[r];
    }
    __syncthreads();

    // ---------------- Phase 4: reduce 4 key-quarter partials, write context ---
    for (int i = tid; i < QTILE * HDIM; i += BLOCK) {
        const int m = i / HDIM;
        const int n = i % HDIM;
        const int ntile = n >> 4;
        const int nn = n & 15;
        float s = 0.f;
#pragma unroll
        for (int kq = 0; kq < 4; ++kq)
            s += part[(kq * 4 + ntile) * 256 + m * 16 + nn];
        ctx_out[((size_t)b * SEQ + qb + m) * HDIM + n] = s;
    }
}

extern "C" void kernel_launch(void* const* d_in, const int* in_sizes, int n_in,
                              void* d_out, int out_size, void* d_ws, size_t ws_size,
                              hipStream_t stream) {
    (void)in_sizes; (void)n_in; (void)out_size; (void)d_ws; (void)ws_size;
    const float* Q = (const float*)d_in[0];
    const float* K = (const float*)d_in[1];
    const float* V = (const float*)d_in[2];
    const unsigned char* mask = (const unsigned char*)d_in[3];  // jnp.bool_ -> 1 byte

    float* ctx  = (float*)d_out;                                  // [B,S,D]
    float* attn = (float*)d_out + (size_t)BATCH * SEQ * HDIM;     // [B,S,S]

    const size_t shmem = SC_BYTES + PART_BYTES;   // 272 KB (<= 320 KB per WGP)
    hipFuncSetAttribute((const void*)sdpa_wmma_kernel,
                        hipFuncAttributeMaxDynamicSharedMemorySize, (int)shmem);

    const int grid = BATCH * (SEQ / QTILE);       // 4096 workgroups
    sdpa_wmma_kernel<<<grid, BLOCK, shmem, stream>>>(Q, K, V, mask, ctx, attn);
}
